// MultiHeadSelfAttention_11390253269036
// MI455X (gfx1250) — matrix operand order repaired and checked by reading, not compiled, not run on hardware
//
#include <hip/hip_runtime.h>

#if __has_builtin(__builtin_amdgcn_global_load_async_to_lds_b128) && \
    __has_builtin(__builtin_amdgcn_s_wait_asynccnt)
#define HAVE_ASYNC 1
#else
#define HAVE_ASYNC 0
#endif

typedef _Float16 half_t;
typedef __attribute__((ext_vector_type(16))) _Float16 v16h;
typedef __attribute__((ext_vector_type(8)))  _Float16 v8h;
typedef __attribute__((ext_vector_type(4)))  _Float16 v4h;
typedef __attribute__((ext_vector_type(8)))  float    v8f;
typedef __attribute__((ext_vector_type(4)))  int      v4i;

// Address-space-qualified 128-bit granule pointers for the async-to-LDS builtin
// (clang prints AS1 as __device__, AS3 as __shared__ in HIP diagnostics).
typedef __attribute__((address_space(1))) v4i* as1_v4i_p;
typedef __attribute__((address_space(3))) v4i* as3_v4i_p;

#define EMBED 512
#define HEADS 8
#define HDIM  64
#define BATCH 4
#define SEQ   2048
#define BS    (BATCH * SEQ)
// (1/sqrt(HDIM)) * log2(e): folded into Q so softmax uses raw exp2 (v_exp_f32)
#define QSCALE 0.18033688011112042f
#define KVS 72     // padded LDS row stride (halves) for K/V/P tiles

__device__ __forceinline__ v8f wmma_f16(v16h a, v16h b, v8f c) {
    return __builtin_amdgcn_wmma_f32_16x16x32_f16(false, a, false, b, (short)0, c,
                                                  false, false);
}

// A fragment (16x32 f16): lane&15 = row M; chunks K = k0+hs*8..+7 and k0+16+hs*8..+7
__device__ __forceinline__ v16h load_a16(const half_t* base, int stride, int m0,
                                         int k0, int lane) {
    int row = m0 + (lane & 15);
    int hs  = lane >> 4;
    const half_t* p = base + (size_t)row * stride + k0 + hs * 8;
    v8h lo = *(const v8h*)p;
    v8h hi = *(const v8h*)(p + 16);
    v16h r;
#pragma unroll
    for (int i = 0; i < 8; ++i) { r[i] = lo[i]; r[i + 8] = hi[i]; }
    return r;
}

// B fragment (32x16 f16) from a K-contiguous (transposed) source:
// lane&15 = column N; chunks K = k0+hs*8..+7 and k0+16+hs*8..+7 (two 16B loads).
__device__ __forceinline__ v16h load_b16(const half_t* base, int stride, int n0,
                                         int k0, int lane) {
    int col = n0 + (lane & 15);
    const half_t* p = base + (size_t)col * stride + k0 + (lane >> 4) * 8;
    v8h lo = *(const v8h*)p;
    v8h hi = *(const v8h*)(p + 16);
    v16h r;
#pragma unroll
    for (int i = 0; i < 8; ++i) { r[i] = lo[i]; r[i + 8] = hi[i]; }
    return r;
}

// 16B global -> LDS copy: async (ASYNCcnt) on toolchains that expose the
// gfx1250 builtin, synchronous VGPR-mediated copy otherwise.
__device__ __forceinline__ void stage16(const half_t* g, half_t* l) {
#if HAVE_ASYNC
    __builtin_amdgcn_global_load_async_to_lds_b128(
        (as1_v4i_p)(void*)g, (as3_v4i_p)(void*)l, 0, 0);
#else
    *(v8h*)l = *(const v8h*)g;
#endif
}

__device__ __forceinline__ void wait_async_pending(bool keep4) {
#if HAVE_ASYNC
    if (keep4) __builtin_amdgcn_s_wait_asynccnt(4);
    else       __builtin_amdgcn_s_wait_asynccnt(0);
#else
    (void)keep4;
#endif
}

__global__ void __launch_bounds__(256) k_cvt_x(const float* __restrict__ x,
                                               half_t* __restrict__ xh, int n4) {
    int i = blockIdx.x * 256 + threadIdx.x;
    if (i >= n4) return;
    float4 v = reinterpret_cast<const float4*>(x)[i];
    v4h o;
    o.x = (half_t)v.x; o.y = (half_t)v.y; o.z = (half_t)v.z; o.w = (half_t)v.w;
    reinterpret_cast<v4h*>(xh)[i] = o;
}

// wT[w][n][k] = W_w[k][n]  (f16, K-contiguous per output column)
__global__ void __launch_bounds__(256) k_cvt_w(const float* __restrict__ Wq,
                                               const float* __restrict__ Wk,
                                               const float* __restrict__ Wv,
                                               const float* __restrict__ Wo,
                                               half_t* __restrict__ wT) {
    int i = blockIdx.x * 256 + threadIdx.x;   // exactly 4*512*512 threads
    int w = i >> 18;
    int n = (i >> 9) & (EMBED - 1);
    int k = i & (EMBED - 1);
    const float* W = (w == 0) ? Wq : (w == 1) ? Wk : (w == 2) ? Wv : Wo;
    wT[i] = (half_t)W[k * EMBED + n];
}

// Fused QKV projection, software-pipelined k-loop (prefetch next fragments
// into registers before issuing current WMMAs).
__global__ void __launch_bounds__(128) k_qkv(const half_t* __restrict__ xh,
                                             const half_t* __restrict__ wT,
                                             const float* __restrict__ bq,
                                             const float* __restrict__ bk,
                                             const float* __restrict__ bv,
                                             half_t* __restrict__ Qh,
                                             half_t* __restrict__ Kh,
                                             half_t* __restrict__ VTh) {
    int lane = threadIdx.x & 31;
    int wid  = blockIdx.x * 4 + (threadIdx.x >> 5);
    int mt   = wid & 511;
    int rest = wid >> 9;
    int head = rest & 7;
    int mat  = rest >> 3;            // 0=Q 1=K 2=V
    int m0   = mt * 16;

    const half_t* Bw   = wT + (size_t)mat * EMBED * EMBED;
    const float*  bias = (mat == 0) ? bq : (mat == 1) ? bk : bv;

    v8f acc[4] = {};
    v16h a = load_a16(xh, EMBED, m0, 0, lane);
    v16h b[4];
#pragma unroll
    for (int nt = 0; nt < 4; ++nt)
        b[nt] = load_b16(Bw, EMBED, head * HDIM + nt * 16, 0, lane);

#pragma unroll
    for (int k0 = 32; k0 < EMBED; k0 += 32) {
        v16h an = load_a16(xh, EMBED, m0, k0, lane);
        v16h bn[4];
#pragma unroll
        for (int nt = 0; nt < 4; ++nt)
            bn[nt] = load_b16(Bw, EMBED, head * HDIM + nt * 16, k0, lane);
        if (k0 + 64 < EMBED)  // warm L0/L2 ahead of the pipeline
            __builtin_prefetch(xh + (size_t)(m0 + (lane & 15)) * EMBED + k0 + 64, 0, 1);
#pragma unroll
        for (int nt = 0; nt < 4; ++nt) acc[nt] = wmma_f16(a, b[nt], acc[nt]);
        a = an;
#pragma unroll
        for (int nt = 0; nt < 4; ++nt) b[nt] = bn[nt];
    }
#pragma unroll
    for (int nt = 0; nt < 4; ++nt) acc[nt] = wmma_f16(a, b[nt], acc[nt]);

    int ncl = lane & 15, hs = lane >> 4;
    int bidx  = m0 >> 11;
    int sbase = (m0 & (SEQ - 1)) + hs * 8;
    int bh    = bidx * HEADS + head;

#pragma unroll
    for (int nt = 0; nt < 4; ++nt) {
        int d = nt * 16 + ncl;
        float bb = bias[head * HDIM + d];
        if (mat == 2) {
            v8h pk;
#pragma unroll
            for (int r = 0; r < 8; ++r) pk[r] = (half_t)(acc[nt][r] + bb);
            *reinterpret_cast<v8h*>(VTh + ((size_t)bh * HDIM + d) * SEQ + sbase) = pk;
        } else {
            half_t* dst = (mat == 0) ? Qh : Kh;
            float scl   = (mat == 0) ? QSCALE : 1.0f;
#pragma unroll
            for (int r = 0; r < 8; ++r)
                dst[((size_t)bh * SEQ + sbase + r) * HDIM + d] =
                    (half_t)((acc[nt][r] + bb) * scl);
        }
    }
}

// Cooperative K/V tile staging: 64 rows x 32 halves each for K and V(transposed),
// 8 KiB per block, 4 async b128 issues per wave per stage.
__device__ __forceinline__ void stage_kv(const half_t* Kp, const half_t* Vp,
                                         half_t* kb, half_t* vb, int n0, int tid) {
    int r0 = tid >> 2, c0 = (tid & 3) * 8;
    stage16(Kp + (size_t)(n0 + r0) * HDIM + c0, kb + r0 * KVS + c0);
    stage16(Vp + (size_t)r0 * SEQ + n0 + c0,    vb + r0 * KVS + c0);
    int t1 = tid + 128;
    int r1 = t1 >> 2, c1 = (t1 & 3) * 8;
    stage16(Kp + (size_t)(n0 + r1) * HDIM + c1, kb + r1 * KVS + c1);
    stage16(Vp + (size_t)r1 * SEQ + n0 + c1,    vb + r1 * KVS + c1);
}

// Flash attention: 4 waves/block share one (b,h); K/V staged to LDS with
// double buffering; per-wave 16 queries, online softmax in exp2 domain.
__global__ void __launch_bounds__(128) k_attn(const half_t* __restrict__ Qh,
                                              const half_t* __restrict__ Kh,
                                              const half_t* __restrict__ VTh,
                                              half_t* __restrict__ Oh) {
    __shared__ __align__(16) half_t Kbuf[2][64 * KVS];
    __shared__ __align__(16) half_t Vbuf[2][64 * KVS];
    __shared__ __align__(16) half_t Ptile[4][16 * KVS];

    int tid  = threadIdx.x;
    int lane = tid & 31;
    int wsl  = tid >> 5;
    int wid  = blockIdx.x * 4 + wsl;
    int mt   = wid & 127;
    int bh   = wid >> 7;            // all 4 waves of a block share bh
    int m0   = mt * 16;

    const half_t* Qp = Qh + (size_t)bh * SEQ * HDIM;
    const half_t* Kp = Kh + (size_t)bh * SEQ * HDIM;
    const half_t* Vp = VTh + (size_t)bh * HDIM * SEQ;
    half_t* Pt = &Ptile[wsl][0];

    v16h q0 = load_a16(Qp, HDIM, m0, 0, lane);
    v16h q1 = load_a16(Qp, HDIM, m0, 32, lane);

    v8f o[4] = {};
    float mrow[8], lrow[8];
#pragma unroll
    for (int r = 0; r < 8; ++r) { mrow[r] = -1e30f; lrow[r] = 0.0f; }

    int nrow = lane & 15, hs = lane >> 4;

    stage_kv(Kp, Vp, Kbuf[0], Vbuf[0], 0, tid);

    for (int n0 = 0; n0 < SEQ; n0 += 64) {
        int  cur  = (n0 >> 6) & 1;
        bool more = (n0 + 64) < SEQ;
        if (more) stage_kv(Kp, Vp, Kbuf[cur ^ 1], Vbuf[cur ^ 1], n0 + 64, tid);
        wait_async_pending(more);   // own 4 oldest async ops (cur buffer) done
        __syncthreads();            // all waves' staging visible

        const half_t* kb = Kbuf[cur];
        const half_t* vb = Vbuf[cur];

        v8f s[4];
#pragma unroll
        for (int t = 0; t < 4; ++t) {
            v16h kf0 = load_b16(kb, KVS, t * 16, 0, lane);
            v16h kf1 = load_b16(kb, KVS, t * 16, 32, lane);
            v8f sa = {};
            sa = wmma_f16(q0, kf0, sa);
            sa = wmma_f16(q1, kf1, sa);
            s[t] = sa;
        }

        float sc[8];
#pragma unroll
        for (int r = 0; r < 8; ++r) {
            float v = fmaxf(fmaxf(s[0][r], s[1][r]), fmaxf(s[2][r], s[3][r]));
            v = fmaxf(v, __shfl_xor(v, 1, 32));
            v = fmaxf(v, __shfl_xor(v, 2, 32));
            v = fmaxf(v, __shfl_xor(v, 4, 32));
            v = fmaxf(v, __shfl_xor(v, 8, 32));
            float mn = fmaxf(mrow[r], v);
            sc[r]   = __builtin_exp2f(mrow[r] - mn);
            mrow[r] = mn;
        }
        float rs[8] = {};
#pragma unroll
        for (int t = 0; t < 4; ++t) {
#pragma unroll
            for (int r = 0; r < 8; ++r) {
                float p = __builtin_exp2f(s[t][r] - mrow[r]);
                rs[r] += p;
                Pt[(hs * 8 + r) * KVS + t * 16 + nrow] = (half_t)p;
            }
        }
#pragma unroll
        for (int r = 0; r < 8; ++r) {
            float x = rs[r];
            x += __shfl_xor(x, 1, 32);
            x += __shfl_xor(x, 2, 32);
            x += __shfl_xor(x, 4, 32);
            x += __shfl_xor(x, 8, 32);
            lrow[r] = lrow[r] * sc[r] + x;
        }
#pragma unroll
        for (int dt = 0; dt < 4; ++dt)
#pragma unroll
            for (int r = 0; r < 8; ++r) o[dt][r] *= sc[r];

        // P: LDS round-trip converts C-layout -> A-layout (wave-local, DScnt order)
        v16h p0 = load_a16(Pt, KVS, 0, 0, lane);
        v16h p1 = load_a16(Pt, KVS, 0, 32, lane);
#pragma unroll
        for (int dt = 0; dt < 4; ++dt) {
            v16h vf0 = load_b16(vb, KVS, dt * 16, 0, lane);
            v16h vf1 = load_b16(vb, KVS, dt * 16, 32, lane);
            o[dt] = wmma_f16(p0, vf0, o[dt]);
            o[dt] = wmma_f16(p1, vf1, o[dt]);
        }
        __syncthreads();            // everyone done with buf[cur] before reuse
    }

    float inv[8];
#pragma unroll
    for (int r = 0; r < 8; ++r) inv[r] = 1.0f / lrow[r];

    int b = bh >> 3, h = bh & 7;
#pragma unroll
    for (int dt = 0; dt < 4; ++dt) {
        int e = h * HDIM + dt * 16 + nrow;
#pragma unroll
        for (int r = 0; r < 8; ++r) {
            int srow = m0 + hs * 8 + r;
            Oh[((size_t)b * SEQ + srow) * EMBED + e] = (half_t)(o[dt][r] * inv[r]);
        }
    }
}

// Output projection: out = Oh @ Wo + bo (f32 result), pipelined like k_qkv.
__global__ void __launch_bounds__(128) k_oproj(const half_t* __restrict__ Oh,
                                               const half_t* __restrict__ woT,
                                               const float* __restrict__ bo,
                                               float* __restrict__ out) {
    int lane = threadIdx.x & 31;
    int wid  = blockIdx.x * 4 + (threadIdx.x >> 5);
    int mt   = wid & 511;
    int ng   = wid >> 9;
    int m0 = mt * 16, n0 = ng * 64;

    v8f acc[4] = {};
    v16h a = load_a16(Oh, EMBED, m0, 0, lane);
    v16h b[4];
#pragma unroll
    for (int nt = 0; nt < 4; ++nt)
        b[nt] = load_b16(woT, EMBED, n0 + nt * 16, 0, lane);

#pragma unroll
    for (int k0 = 32; k0 < EMBED; k0 += 32) {
        v16h an = load_a16(Oh, EMBED, m0, k0, lane);
        v16h bn[4];
#pragma unroll
        for (int nt = 0; nt < 4; ++nt)
            bn[nt] = load_b16(woT, EMBED, n0 + nt * 16, k0, lane);
#pragma unroll
        for (int nt = 0; nt < 4; ++nt) acc[nt] = wmma_f16(a, b[nt], acc[nt]);
        a = an;
#pragma unroll
        for (int nt = 0; nt < 4; ++nt) b[nt] = bn[nt];
    }
#pragma unroll
    for (int nt = 0; nt < 4; ++nt) acc[nt] = wmma_f16(a, b[nt], acc[nt]);

    int ncl = lane & 15, hs = lane >> 4;
#pragma unroll
    for (int nt = 0; nt < 4; ++nt) {
        int n = n0 + nt * 16 + ncl;
        float bb = bo[n];
#pragma unroll
        for (int r = 0; r < 8; ++r)
            out[(size_t)(m0 + hs * 8 + r) * EMBED + n] = acc[nt][r] + bb;
    }
}

extern "C" void kernel_launch(void* const* d_in, const int* in_sizes, int n_in,
                              void* d_out, int out_size, void* d_ws, size_t ws_size,
                              hipStream_t stream) {
    const float* x  = (const float*)d_in[0];
    const float* Wq = (const float*)d_in[1];
    const float* bq = (const float*)d_in[2];
    const float* Wk = (const float*)d_in[3];
    const float* bk = (const float*)d_in[4];
    const float* Wv = (const float*)d_in[5];
    const float* bv = (const float*)d_in[6];
    const float* Wo = (const float*)d_in[7];
    const float* bo = (const float*)d_in[8];

    char* ws = (char*)d_ws;
    half_t* xh  = (half_t*)(ws);                 // 8192*512  f16 =  8 MiB
    half_t* wT  = (half_t*)(ws + 8388608);       // 4*512*512 f16 =  2 MiB
    half_t* Qh  = (half_t*)(ws + 10485760);      // [bh][s][d]    =  8 MiB
    half_t* Kh  = (half_t*)(ws + 18874368);      // [bh][s][d]    =  8 MiB
    half_t* VTh = (half_t*)(ws + 27262976);      // [bh][d][s]    =  8 MiB
    half_t* Oh  = (half_t*)(ws + 35651584);      // [s][e]        =  8 MiB

    k_cvt_x<<<4096, 256, 0, stream>>>(x, xh, BS * EMBED / 4);
    k_cvt_w<<<4096, 256, 0, stream>>>(Wq, Wk, Wv, Wo, wT);
    k_qkv<<<3072, 128, 0, stream>>>(xh, wT, bq, bk, bv, Qh, Kh, VTh);
    k_attn<<<1024, 128, 0, stream>>>(Qh, Kh, VTh, Oh);
    k_oproj<<<1024, 128, 0, stream>>>(Oh, wT + 3 * EMBED * EMBED, bo, (float*)d_out);
}
